// TransformerLayer_48232482734148
// MI455X (gfx1250) — matrix operand order repaired and checked by reading, not compiled, not run on hardware
//
#include <hip/hip_runtime.h>
#include <hip/hip_bf16.h>
#include <math.h>

typedef __bf16 bf16_t;
typedef bf16_t v16bf __attribute__((ext_vector_type(16)));
typedef bf16_t v8bf  __attribute__((ext_vector_type(8)));
typedef float  v8f   __attribute__((ext_vector_type(8)));

namespace {

constexpr int kB = 2, kS = 2048, kD = 1024, kH = 16, kF = 4096, kE = 8;
constexpr int kDH = kD / kH;     // 64
constexpr int kM  = kB * kS;     // 4096 tokens

// LDS row pitch for staged B tiles: 128 cols * 2B = 256B + 16B pad = 272B.
// Per-lane 16B reads then start 4 banks apart per row -> 2-way (the floor for
// a 512B/instruction request), instead of fully serialized at 256B pitch.
constexpr int kBPitch = 136;     // elements (272 bytes)

// ---------------------------------------------------------------- WMMA core
__device__ __forceinline__ v8f wmma_bf16(v16bf a, v16bf b, v8f c) {
  // v_wmma_f32_16x16x32_bf16: D = A(16x32) x B(32x16) + C(16x16 f32)
  return __builtin_amdgcn_wmma_f32_16x16x32_bf16(false, a, false, b,
                                                 (short)0, c, false, false);
}

// A fragment: 16 rows x 32 K, bf16 row-major source, ld in elements.
// lane<16: row=lane, K={0..7,16..23}; lane>=16: row=lane-16, K={8..15,24..31}
__device__ __forceinline__ v16bf load_fragA(const bf16_t* base, int ld) {
  const int lane = threadIdx.x & 31;
  const int r    = lane & 15;
  const int koff = (lane >> 4) << 3;           // 0 or 8
  const bf16_t* p = base + (size_t)r * ld + koff;
  union { v16bf v; v8bf h[2]; } u;
  u.h[0] = *(const v8bf*)(p);
  u.h[1] = *(const v8bf*)(p + 16);
  return u.v;
}

// B fragment: 32 K x 16 N, bf16 row-major [K][N] source (lane = N column,
// lane<16: K={0..7,16..23}; lane>=16: K={8..15,24..31}).  ld in elements.
__device__ __forceinline__ v16bf load_fragB(const bf16_t* base, int ld) {
  const int lane = threadIdx.x & 31;
  const int koff = (lane >> 4) << 3;           // 0 or 8
  const bf16_t* p = base + (size_t)koff * ld + (lane & 15);
  v16bf v;
#pragma unroll
  for (int s = 0; s < 8; ++s) {
    v[s]     = p[(size_t)s * ld];
    v[8 + s] = p[(size_t)(16 + s) * ld];
  }
  return v;
}

__device__ __forceinline__ float gelu_tanh(float x) {
  const float c = 0.7978845608028654f;         // sqrt(2/pi)
  float t = tanhf(c * (x + 0.044715f * x * x * x));
  return 0.5f * x * (1.0f + t);
}

// ---- CDNA5 async copy: global -> LDS, ASYNCcnt-tracked (no VGPR round-trip)
__device__ __forceinline__ void async_g2l_b128(unsigned lds_byte_off,
                                               const bf16_t* gaddr) {
  asm volatile("global_load_async_to_lds_b128 %0, %1, off"
               :: "v"(lds_byte_off), "v"(gaddr) : "memory");
}
__device__ __forceinline__ void wait_async0() {
  asm volatile("s_wait_asynccnt 0x0" ::: "memory");
}
// Truncate a generic pointer to an LDS byte offset (LDS aperture keeps the
// allocation-relative offset in addr[31:0]).
__device__ __forceinline__ unsigned lds_off_of(const void* p) {
  return (unsigned)(size_t)p;
}

// Stage one 32(K) x 128(N) bf16 B tile into LDS: 512 x 16B vectors, 2/thread.
__device__ __forceinline__ void stage_b_async(const bf16_t* gbase, int ldN,
                                              unsigned lds_base) {
  const int t = threadIdx.x;
#pragma unroll
  for (int i = 0; i < 2; ++i) {
    const int v   = t + i * 256;       // 0..511
    const int row = v >> 4;            // 0..31
    const int c8  = (v & 15) << 3;     // element offset within row
    async_g2l_b128(lds_base + (unsigned)(row * (kBPitch * 2) + c8 * 2),
                   gbase + (size_t)row * ldN + c8);
  }
}

// ---------------------------------------------------------------- converters
__global__ void cvt_f32_bf16_kernel(const float* __restrict__ in,
                                    bf16_t* __restrict__ out, size_t n) {
  size_t i = (size_t)blockIdx.x * blockDim.x + threadIdx.x;
  if (i < n) out[i] = (bf16_t)in[i];
}

// K projection output [B,S,D] -> per-head transposed Kt [B,H,DH,S]
__global__ void head_transpose_kernel(const bf16_t* __restrict__ in,
                                      bf16_t* __restrict__ out) {
  size_t i = (size_t)blockIdx.x * blockDim.x + threadIdx.x;
  if (i >= (size_t)kM * kD) return;
  int c  = (int)(i % kD);
  size_t tok = i / kD;
  int b = (int)(tok / kS), s = (int)(tok % kS);
  int h = c >> 6, dh = c & 63;
  out[(((size_t)(b * kH + h)) * kDH + dh) * kS + s] = in[i];
}

// ---------------------------------------------------------------- GEMM
enum { EPI_BIAS = 0, EPI_GELU = 1, EPI_MOE = 2 };

// C[M,N] = A[M,K](bf16) x B[K,N](bf16, row-major) + bias[N]; epilogue variants.
// Block: 256 threads = 8 waves (4 row x 2 col), block tile 128x128,
// wave tile 32x64 (2 A-frags x 4 B-frags = 8 WMMA / K-step).
// B tiles are double-buffered in LDS via async global->LDS copies.
template <int EPI, bool BF16OUT>
__global__ __launch_bounds__(256) void gemm_kernel(
    const bf16_t* __restrict__ A, const bf16_t* __restrict__ Bw,
    const float* __restrict__ bias, float* __restrict__ Cf,
    bf16_t* __restrict__ Cb, int M, int N, int K,
    const float* __restrict__ gate, int expert) {
  __shared__ __align__(16) bf16_t bsm[2][32 * kBPitch];
  const int wave = threadIdx.x >> 5;
  const int lane = threadIdx.x & 31;
  const int wm = wave & 3, wn = wave >> 2;
  const int row0 = blockIdx.y * 128 + wm * 32;
  const int colB = blockIdx.x * 128;           // block column base
  const int col0 = colB + wn * 64;             // wave column base

  const unsigned lds0 = lds_off_of(&bsm[0][0]);
  const unsigned lds1 = lds_off_of(&bsm[1][0]);

  v8f c[2][4] = {};
  const bf16_t* A0 = A + (size_t)row0 * K;

  stage_b_async(Bw + colB, N, lds0);           // prefetch tile kb=0
  for (int kb = 0; kb < K; kb += 32) {
    const int cur = (kb >> 5) & 1;
    wait_async0();
    __syncthreads();                           // staged tile visible, prev reads done
    if (kb + 32 < K)
      stage_b_async(Bw + (size_t)(kb + 32) * N + colB, N, cur ? lds0 : lds1);

    v16bf a0 = load_fragA(A0 + kb, K);
    v16bf a1 = load_fragA(A0 + (size_t)16 * K + kb, K);
    const bf16_t* bt = &bsm[cur][0] + wn * 64;
#pragma unroll
    for (int t = 0; t < 4; ++t) {
      v16bf b = load_fragB(bt + t * 16, kBPitch);
      c[0][t] = wmma_bf16(a0, b, c[0][t]);
      c[1][t] = wmma_bf16(a1, b, c[1][t]);
    }
  }

  const int nl   = lane & 15;
  const int mtop = (lane >> 4) << 3;           // 0 or 8
#pragma unroll
  for (int rt = 0; rt < 2; ++rt) {
#pragma unroll
    for (int t = 0; t < 4; ++t) {
      const int col = col0 + t * 16 + nl;
      const float bv = bias[col];
#pragma unroll
      for (int r = 0; r < 8; ++r) {
        const int m = row0 + rt * 16 + mtop + r;
        float val = c[rt][t][r] + bv;
        if (EPI == EPI_GELU) val = gelu_tanh(val);
        const size_t idx = (size_t)m * N + col;
        if (EPI == EPI_MOE) {
          Cf[idx] += gate[(size_t)m * kE + expert] * val;
        } else if (BF16OUT) {
          Cb[idx] = (bf16_t)val;
        } else {
          Cf[idx] = val;
        }
      }
    }
  }
}

// ---------------------------------------------------------------- attention
// Flash attention, one wave per (b, h, 16-query tile); 8 waves / block.
__global__ __launch_bounds__(256) void attn_kernel(
    const bf16_t* __restrict__ Q, const bf16_t* __restrict__ Kt,
    const bf16_t* __restrict__ V, const float* __restrict__ mask,
    bf16_t* __restrict__ ctx) {
  __shared__ __align__(16) bf16_t plds[8][16 * 32];
  const int wavei = threadIdx.x >> 5;
  const int lane  = threadIdx.x & 31;
  const int wid = blockIdx.x * 8 + wavei;
  const int qt = wid & (kS / 16 - 1);          // 128 q-tiles
  const int h  = (wid >> 7) & (kH - 1);
  const int b  = wid >> 11;
  const int q0 = qt * 16;

  const bf16_t* qbase = Q + ((size_t)b * kS + q0) * kD + h * kDH;
  const bf16_t* ktb   = Kt + ((size_t)(b * kH + h)) * kDH * kS;   // [DH][S]
  const bf16_t* vbase = V + (size_t)b * kS * kD + h * kDH;        // rows = keys
  const float*  mrow  = mask + ((size_t)b * kS + q0) * kS;        // [q][k]
  bf16_t* myp = plds[wavei];

  const v16bf qa0 = load_fragA(qbase, kD);       // dh 0..31
  const v16bf qa1 = load_fragA(qbase + 32, kD);  // dh 32..63

  float mrun[8], lrun[8];
  v8f o[4] = {};
#pragma unroll
  for (int r = 0; r < 8; ++r) { mrun[r] = -INFINITY; lrun[r] = 0.f; }

  const int nl   = lane & 15;
  const int mtop = (lane >> 4) << 3;

  for (int kb = 0; kb < kS; kb += 32) {
    // ---- scores: 16q x 32keys (two 16x16 fragments), K-dim = DH = 64
    v8f sc[2] = {};
#pragma unroll
    for (int t = 0; t < 2; ++t) {
      sc[t] = wmma_bf16(qa0, load_fragB(ktb + (kb + t * 16), kS), sc[t]);
      sc[t] = wmma_bf16(qa1, load_fragB(ktb + (size_t)32 * kS + kb + t * 16, kS), sc[t]);
    }
    // ---- scale + additive mask
#pragma unroll
    for (int t = 0; t < 2; ++t)
#pragma unroll
      for (int r = 0; r < 8; ++r)
        sc[t][r] = sc[t][r] * 0.125f +
                   mrow[(size_t)(mtop + r) * kS + kb + t * 16 + nl];
    // ---- online softmax (row r lives in one 16-lane half)
    float alpha[8];
#pragma unroll
    for (int r = 0; r < 8; ++r) {
      float mx = fmaxf(sc[0][r], sc[1][r]);
#pragma unroll
      for (int off = 8; off >= 1; off >>= 1) mx = fmaxf(mx, __shfl_xor(mx, off));
      const float mn = fmaxf(mrun[r], mx);
      alpha[r] = __expf(mrun[r] - mn);
      mrun[r] = mn;
      const float p0 = __expf(sc[0][r] - mn);
      const float p1 = __expf(sc[1][r] - mn);
      sc[0][r] = p0; sc[1][r] = p1;
      float ps = p0 + p1;
#pragma unroll
      for (int off = 8; off >= 1; off >>= 1) ps += __shfl_xor(ps, off);
      lrun[r] = lrun[r] * alpha[r] + ps;
    }
#pragma unroll
    for (int t = 0; t < 4; ++t)
#pragma unroll
      for (int r = 0; r < 8; ++r) o[t][r] *= alpha[r];
    // ---- re-layout P (C-frag -> A-frag) through LDS
#pragma unroll
    for (int t = 0; t < 2; ++t)
#pragma unroll
      for (int r = 0; r < 8; ++r)
        myp[(mtop + r) * 32 + t * 16 + nl] = (bf16_t)sc[t][r];
    __builtin_amdgcn_wave_barrier();
    const v16bf pa = load_fragA((const bf16_t*)myp, 32);
    // ---- O += P x V  (B-frag lane = dh column, regs = 16 key rows)
#pragma unroll
    for (int t = 0; t < 4; ++t) {
      v16bf vb = load_fragB(vbase + (size_t)kb * kD + t * 16, kD);
      o[t] = wmma_bf16(pa, vb, o[t]);
    }
  }
  // ---- finalize: divide by running sum, merge heads back to [B,S,D]
  float inv[8];
#pragma unroll
  for (int r = 0; r < 8; ++r) inv[r] = 1.f / lrun[r];
#pragma unroll
  for (int t = 0; t < 4; ++t)
#pragma unroll
    for (int r = 0; r < 8; ++r)
      ctx[((size_t)b * kS + q0 + mtop + r) * kD + h * kDH + t * 16 + nl] =
          (bf16_t)(o[t][r] * inv[r]);
}

// ---------------------------------------------------------------- LayerNorm
__global__ __launch_bounds__(256) void add_ln_kernel(
    const float* __restrict__ x, const float* __restrict__ res,
    const float* __restrict__ g, const float* __restrict__ beta,
    float* __restrict__ outf, bf16_t* __restrict__ outb) {
  const int row = blockIdx.x;
  const size_t base = (size_t)row * kD;
  float v[4], sum = 0.f, sq = 0.f;
#pragma unroll
  for (int j = 0; j < 4; ++j) {
    const int i = threadIdx.x + j * 256;
    const float t = x[base + i] + res[base + i];
    v[j] = t; sum += t; sq += t * t;
  }
#pragma unroll
  for (int off = 16; off >= 1; off >>= 1) {
    sum += __shfl_xor(sum, off);
    sq  += __shfl_xor(sq, off);
  }
  __shared__ float red[16];
  const int wave = threadIdx.x >> 5, lane = threadIdx.x & 31;
  if (lane == 0) { red[wave] = sum; red[8 + wave] = sq; }
  __syncthreads();
  if (threadIdx.x == 0) {
    float s = 0.f, q = 0.f;
    for (int i = 0; i < 8; ++i) { s += red[i]; q += red[8 + i]; }
    red[0] = s; red[8] = q;
  }
  __syncthreads();
  sum = red[0]; sq = red[8];
  const float mu   = sum * (1.f / kD);
  const float var  = sq * (1.f / kD) - mu * mu;
  const float rstd = rsqrtf(var + 1e-6f);
#pragma unroll
  for (int j = 0; j < 4; ++j) {
    const int i = threadIdx.x + j * 256;
    const float y = (v[j] - mu) * rstd * g[i] + beta[i];
    outf[base + i] = y;
    if (outb) outb[base + i] = (bf16_t)y;
  }
}

// ---------------------------------------------------------------- gate
__global__ __launch_bounds__(256) void gate_kernel(
    const float* __restrict__ x, const float* __restrict__ gw,
    const float* __restrict__ gb, float* __restrict__ gate) {
  const int row = blockIdx.x;
  const int wave = threadIdx.x >> 5, lane = threadIdx.x & 31;  // wave = expert
  float acc = 0.f;
  for (int i = lane; i < kD; i += 32)
    acc += x[(size_t)row * kD + i] * gw[(size_t)i * kE + wave];
#pragma unroll
  for (int off = 16; off >= 1; off >>= 1) acc += __shfl_xor(acc, off);
  __shared__ float lg[kE];
  if (lane == 0) lg[wave] = acc + gb[wave];
  __syncthreads();
  if (threadIdx.x == 0) {
    float mx = lg[0];
    for (int e = 1; e < kE; ++e) mx = fmaxf(mx, lg[e]);
    float ex[kE], s = 0.f;
    for (int e = 0; e < kE; ++e) { ex[e] = __expf(lg[e] - mx); s += ex[e]; }
    const float inv = 1.f / s;
    for (int e = 0; e < kE; ++e) gate[(size_t)row * kE + e] = ex[e] * inv;
  }
}

}  // namespace

// ---------------------------------------------------------------- launcher
extern "C" void kernel_launch(void* const* d_in, const int* in_sizes, int n_in,
                              void* d_out, int out_size, void* d_ws,
                              size_t ws_size, hipStream_t stream) {
  (void)in_sizes; (void)n_in; (void)out_size; (void)ws_size;
  const float* hidden = (const float*)d_in[0];
  const float* mask   = (const float*)d_in[1];
  const float* wq = (const float*)d_in[2];  const float* bq = (const float*)d_in[3];
  const float* wk = (const float*)d_in[4];  const float* bk = (const float*)d_in[5];
  const float* wv = (const float*)d_in[6];  const float* bv = (const float*)d_in[7];
  const float* wo = (const float*)d_in[8];  const float* bo = (const float*)d_in[9];
  const float* ln1_g = (const float*)d_in[10]; const float* ln1_b = (const float*)d_in[11];
  const float* gate_w = (const float*)d_in[12]; const float* gate_b = (const float*)d_in[13];
  const float* w1 = (const float*)d_in[14]; const float* b1 = (const float*)d_in[15];
  const float* w2 = (const float*)d_in[16]; const float* b2 = (const float*)d_in[17];
  const float* ln2_g = (const float*)d_in[18]; const float* ln2_b = (const float*)d_in[19];

  float* out      = (float*)d_out;
  float* gate_out = out + (size_t)kM * kD;   // tuple part 2: [B,S,E]

  // ---- workspace carve (256B aligned)
  char* p = (char*)d_ws;
  auto carve = [&](size_t bytes) -> char* {
    char* r = p; p += (bytes + 255) & ~(size_t)255; return r;
  };
  const size_t nTD = (size_t)kM * kD;        // tokens x D
  bf16_t* wq_b  = (bf16_t*)carve((size_t)kD * kD * 2);
  bf16_t* wk_b  = (bf16_t*)carve((size_t)kD * kD * 2);
  bf16_t* wv_b  = (bf16_t*)carve((size_t)kD * kD * 2);
  bf16_t* wo_b  = (bf16_t*)carve((size_t)kD * kD * 2);
  bf16_t* w1_b  = (bf16_t*)carve((size_t)kE * kD * kF * 2);
  bf16_t* w2_b  = (bf16_t*)carve((size_t)kE * kF * kD * 2);
  bf16_t* hid_b = (bf16_t*)carve(nTD * 2);
  bf16_t* q_b   = (bf16_t*)carve(nTD * 2);
  bf16_t* k_b   = (bf16_t*)carve(nTD * 2);
  bf16_t* v_b   = (bf16_t*)carve(nTD * 2);
  bf16_t* kt_b  = (bf16_t*)carve(nTD * 2);
  bf16_t* ctx_b = (bf16_t*)carve(nTD * 2);
  float*  attn_f  = (float*)carve(nTD * 4);
  float*  inter_f = (float*)carve(nTD * 4);
  bf16_t* inter_b = (bf16_t*)carve(nTD * 2);
  bf16_t* h_b     = (bf16_t*)carve((size_t)kM * kF * 2);
  float*  moe_f   = (float*)carve(nTD * 4);

  const int TPB = 256;
  auto blocks = [](size_t n) { return (unsigned)((n + 255) / 256); };

  // ---- f32 -> bf16 conversions (weights + activations)
  cvt_f32_bf16_kernel<<<blocks(nTD), TPB, 0, stream>>>(hidden, hid_b, nTD);
  cvt_f32_bf16_kernel<<<blocks((size_t)kD * kD), TPB, 0, stream>>>(wq, wq_b, (size_t)kD * kD);
  cvt_f32_bf16_kernel<<<blocks((size_t)kD * kD), TPB, 0, stream>>>(wk, wk_b, (size_t)kD * kD);
  cvt_f32_bf16_kernel<<<blocks((size_t)kD * kD), TPB, 0, stream>>>(wv, wv_b, (size_t)kD * kD);
  cvt_f32_bf16_kernel<<<blocks((size_t)kD * kD), TPB, 0, stream>>>(wo, wo_b, (size_t)kD * kD);
  cvt_f32_bf16_kernel<<<blocks((size_t)kE * kD * kF), TPB, 0, stream>>>(w1, w1_b, (size_t)kE * kD * kF);
  cvt_f32_bf16_kernel<<<blocks((size_t)kE * kF * kD), TPB, 0, stream>>>(w2, w2_b, (size_t)kE * kF * kD);

  // ---- Q,K,V projections (bf16 WMMA GEMMs, bias epilogue, bf16 out)
  const dim3 gDD(kD / 128, kM / 128);
  gemm_kernel<EPI_BIAS, true><<<gDD, TPB, 0, stream>>>(hid_b, wq_b, bq, nullptr, q_b, kM, kD, kD, nullptr, 0);
  gemm_kernel<EPI_BIAS, true><<<gDD, TPB, 0, stream>>>(hid_b, wk_b, bk, nullptr, k_b, kM, kD, kD, nullptr, 0);
  gemm_kernel<EPI_BIAS, true><<<gDD, TPB, 0, stream>>>(hid_b, wv_b, bv, nullptr, v_b, kM, kD, kD, nullptr, 0);

  // ---- per-head transpose of K for score B-fragments
  head_transpose_kernel<<<blocks(nTD), TPB, 0, stream>>>(k_b, kt_b);

  // ---- flash attention (4096 waves, 8 per block)
  attn_kernel<<<kB * kH * (kS / 16) / 8, TPB, 0, stream>>>(q_b, kt_b, v_b, mask, ctx_b);

  // ---- output projection (f32 out)
  gemm_kernel<EPI_BIAS, false><<<gDD, TPB, 0, stream>>>(ctx_b, wo_b, bo, attn_f, nullptr, kM, kD, kD, nullptr, 0);

  // ---- LN1(attn + hidden) -> inter (f32 + bf16)
  add_ln_kernel<<<kM, TPB, 0, stream>>>(attn_f, hidden, ln1_g, ln1_b, inter_f, inter_b);

  // ---- gate softmax -> second output
  gate_kernel<<<kM, TPB, 0, stream>>>(inter_f, gate_w, gate_b, gate_out);

  // ---- soft MoE: sequential experts, gate-weighted accumulation
  hipMemsetAsync(moe_f, 0, nTD * 4, stream);
  const dim3 gDF(kF / 128, kM / 128);
  for (int e = 0; e < kE; ++e) {
    gemm_kernel<EPI_GELU, true><<<gDF, TPB, 0, stream>>>(
        inter_b, w1_b + (size_t)e * kD * kF, b1 + (size_t)e * kF,
        nullptr, h_b, kM, kF, kD, nullptr, 0);
    gemm_kernel<EPI_MOE, false><<<gDD, TPB, 0, stream>>>(
        h_b, w2_b + (size_t)e * kF * kD, b2 + (size_t)e * kD,
        moe_f, nullptr, kM, kD, kF, gate_out, e);
  }

  // ---- LN2(moe + inter) -> first output
  add_ln_kernel<<<kM, TPB, 0, stream>>>(moe_f, inter_f, ln2_g, ln2_b, out, nullptr);
}
